// MoELayer_79422535238245
// MI455X (gfx1250) — matrix operand order repaired and checked by reading, not compiled, not run on hardware
//
#include <hip/hip_runtime.h>
#include <hip/hip_bf16.h>
#include <math.h>

// MoE router for MI455X (gfx1250, wave32).
// GEMMs via v_wmma_f32_16x16x32_bf16 (f32 accumulate). Memory-bound design:
// ~650MB traffic -> ~28us floor at 23.3 TB/s; bf16 WMMA keeps compute far below that.
// Each wave owns 32 tokens (two 16-row A tiles) so every B fragment loaded from LDS
// feeds two WMMAs -> LDS read traffic halved vs. one tile per wave.

typedef __attribute__((ext_vector_type(16))) __bf16 v16bf;
typedef __attribute__((ext_vector_type(8)))  __bf16 v8bf;
typedef __attribute__((ext_vector_type(4)))  __bf16 v4bf;
typedef __attribute__((ext_vector_type(8)))  float  v8f;

#define NTOK 262144
#define KDIM 512
#define NEXP 64
#define DE   64
#define TOK_PER_WG 256   // 8 waves * 32 tokens

__global__ __launch_bounds__(256)
void moe_router_kernel(const float* __restrict__ h, const float* __restrict__ W,
                       const float* __restrict__ E, const float* __restrict__ taup,
                       float* __restrict__ outS, int* __restrict__ outI,
                       float* __restrict__ outF)
{
    __shared__ __align__(32) __bf16 sW[DE * KDIM];     // 64 KB: W as bf16, row-major [d_e][k]
    __shared__ __align__(32) __bf16 sE[NEXP * DE];     //  8 KB: normalized expert emb [e][k]
    __shared__ __align__(32) __bf16 sH[8 * 32 * DE];   // 32 KB: per-wave h_norm tiles [32][64]

    const int tid  = threadIdx.x;
    const int lane = tid & 31;
    const int wave = tid >> 5;
    const int half = lane >> 4;   // 0 or 1 (16-lane half)
    const int ln   = lane & 15;

    // ---- Stage W (f32 -> bf16) into LDS; flat copy preserves [64][512] row-major ----
    for (int i = tid; i < (DE * KDIM) / 4; i += 256) {
        float4 v = reinterpret_cast<const float4*>(W)[i];
        v4bf o;
        o[0] = (__bf16)v.x; o[1] = (__bf16)v.y; o[2] = (__bf16)v.z; o[3] = (__bf16)v.w;
        reinterpret_cast<v4bf*>(sW)[i] = o;
    }
    // ---- Normalize expert embeddings (one row per thread, 64 rows) ----
    if (tid < NEXP) {
        const float* row = E + tid * DE;
        float ss = 0.0f;
        #pragma unroll 8
        for (int j = 0; j < DE; ++j) { float x = row[j]; ss = fmaf(x, x, ss); }
        float inv = 1.0f / fmaxf(sqrtf(ss), 1e-12f);
        #pragma unroll 8
        for (int j = 0; j < DE; ++j) sE[tid * DE + j] = (__bf16)(row[j] * inv);
    }
    __syncthreads();

    // A-fragment k order used here (contiguous; B uses the same): lane (half,ln) holds row M=ln,
    // K = 16*half + [0,16) of the 32-wide k-step.
    const int tokbase = blockIdx.x * TOK_PER_WG + wave * 32;
    const float* hrow0 = h + (size_t)(tokbase + ln) * KDIM;        // A tile u=0
    const float* hrow1 = h + (size_t)(tokbase + 16 + ln) * KDIM;   // A tile u=1

    // -------- GEMM1: h_proj[32 x 64] = h_tile[32 x 512] @ W^T --------
    v8f acc[2][4] = {};
    #pragma unroll 2
    for (int ks = 0; ks < KDIM / 32; ++ks) {
        const int k0 = ks * 32 + half * 16;
        // B fragments for the 4 expert-dim tiles; each feeds both A tiles.
        // B (32x16): lane column n=ln, K = 16*half + [0,16) -> 32 contiguous bytes
        // of W row (t*16+ln).
        v16bf b[4];
        #pragma unroll
        for (int t = 0; t < 4; ++t)
            b[t] = *reinterpret_cast<const v16bf*>(
                &sW[(size_t)(t * 16 + ln) * KDIM + ks * 32 + half * 16]);
        #pragma unroll
        for (int u = 0; u < 2; ++u) {
            const float* hr = u ? hrow1 : hrow0;
            float4 a0 = *reinterpret_cast<const float4*>(hr + k0);
            float4 a1 = *reinterpret_cast<const float4*>(hr + k0 + 4);
            float4 a2 = *reinterpret_cast<const float4*>(hr + k0 + 8);
            float4 a3 = *reinterpret_cast<const float4*>(hr + k0 + 12);
            v16bf a;
            a[0]=(__bf16)a0.x;  a[1]=(__bf16)a0.y;  a[2]=(__bf16)a0.z;  a[3]=(__bf16)a0.w;
            a[4]=(__bf16)a1.x;  a[5]=(__bf16)a1.y;  a[6]=(__bf16)a1.z;  a[7]=(__bf16)a1.w;
            a[8]=(__bf16)a2.x;  a[9]=(__bf16)a2.y;  a[10]=(__bf16)a2.z; a[11]=(__bf16)a2.w;
            a[12]=(__bf16)a3.x; a[13]=(__bf16)a3.y; a[14]=(__bf16)a3.z; a[15]=(__bf16)a3.w;
            #pragma unroll
            for (int t = 0; t < 4; ++t)
                acc[u][t] = __builtin_amdgcn_wmma_f32_16x16x32_bf16(
                    false, a, false, b[t], (short)0, acc[u][t], false, false);
        }
    }

    // -------- Row L2 norm of h_proj; write h_norm tiles (bf16) to LDS --------
    // C/D layout: lane (half,ln) VGPR r holds D[M = r + 8*half][N = ln].
    __bf16* hn = &sH[wave * 32 * DE];
    #pragma unroll
    for (int u = 0; u < 2; ++u) {
        #pragma unroll
        for (int r = 0; r < 8; ++r) {
            float p = 0.0f;
            #pragma unroll
            for (int t = 0; t < 4; ++t) { float v = acc[u][t][r]; p = fmaf(v, v, p); }
            #pragma unroll
            for (int m = 1; m < 16; m <<= 1) p += __shfl_xor(p, m, 32);
            float sc = 1.0f / fmaxf(sqrtf(p), 1e-12f);
            const int mrow = u * 16 + r + 8 * half;
            #pragma unroll
            for (int t = 0; t < 4; ++t)
                hn[mrow * DE + t * 16 + ln] = (__bf16)(acc[u][t][r] * sc);
        }
    }
    __syncthreads();

    const float inv_tau = 1.0f / taup[0];

    // -------- GEMM2 + softmax + top-2, one 16-token tile at a time --------
    #pragma unroll
    for (int u = 0; u < 2; ++u) {
        // scores[16 x 64] = h_norm[16 x 64] @ e_norm^T
        v8f sc4[4] = {};
        #pragma unroll
        for (int ks = 0; ks < 2; ++ks) {
            const int k0 = ks * 32 + half * 16;
            const int arow = (u * 16 + ln) * DE;
            v8bf c0 = *reinterpret_cast<const v8bf*>(&hn[arow + k0]);
            v8bf c1 = *reinterpret_cast<const v8bf*>(&hn[arow + k0 + 8]);
            v16bf a;
            #pragma unroll
            for (int j = 0; j < 8; ++j) { a[j] = c0[j]; a[8 + j] = c1[j]; }
            #pragma unroll
            for (int t = 0; t < 4; ++t) {
                const v16bf b = *reinterpret_cast<const v16bf*>(
                    &sE[(t * 16 + ln) * DE + ks * 32 + half * 16]);
                sc4[t] = __builtin_amdgcn_wmma_f32_16x16x32_bf16(
                    false, a, false, b, (short)0, sc4[t], false, false);
            }
        }

        #pragma unroll
        for (int r = 0; r < 8; ++r) {
            float v[4];
            #pragma unroll
            for (int t = 0; t < 4; ++t) v[t] = sc4[t][r] * inv_tau;
            float mx = fmaxf(fmaxf(v[0], v[1]), fmaxf(v[2], v[3]));
            #pragma unroll
            for (int m = 1; m < 16; m <<= 1) mx = fmaxf(mx, __shfl_xor(mx, m, 32));
            float e[4]; float s = 0.0f;
            #pragma unroll
            for (int t = 0; t < 4; ++t) { e[t] = __expf(v[t] - mx); s += e[t]; }
            #pragma unroll
            for (int m = 1; m < 16; m <<= 1) s += __shfl_xor(s, m, 32);
            const float inv_s = 1.0f / s;
            float g[4];
            #pragma unroll
            for (int t = 0; t < 4; ++t) g[t] = e[t] * inv_s;

            // top-1: max value, then min index among ties (JAX tie-break)
            float bm = g[0]; int bi = ln;
            #pragma unroll
            for (int t = 1; t < 4; ++t) { if (g[t] > bm) { bm = g[t]; bi = t * 16 + ln; } }
            float M1 = bm;
            #pragma unroll
            for (int m = 1; m < 16; m <<= 1) M1 = fmaxf(M1, __shfl_xor(M1, m, 32));
            int c1 = (bm == M1) ? bi : (1 << 20);
            #pragma unroll
            for (int m = 1; m < 16; m <<= 1) c1 = min(c1, __shfl_xor(c1, m, 32));
            // top-2: exclude index c1
            float bm2 = -3.4e38f; int bi2 = (1 << 20);
            #pragma unroll
            for (int t = 0; t < 4; ++t) {
                int id = t * 16 + ln;
                if (id != c1 && g[t] > bm2) { bm2 = g[t]; bi2 = id; }
            }
            float M2 = bm2;
            #pragma unroll
            for (int m = 1; m < 16; m <<= 1) M2 = fmaxf(M2, __shfl_xor(M2, m, 32));
            int c2 = (bm2 == M2) ? bi2 : (1 << 20);
            #pragma unroll
            for (int m = 1; m < 16; m <<= 1) c2 = min(c2, __shfl_xor(c2, m, 32));

            // normalized_topk = softmax([M1, M2]) over the two gate values
            const float ew = __expf(M2 - M1);
            const float w1 = 1.0f / (1.0f + ew);
            const float w2 = ew * w1;

            const int tok = tokbase + u * 16 + r + 8 * half;
            float* frow = outF + (size_t)tok * NEXP;
            float* srow = outS + (size_t)tok * NEXP;
            #pragma unroll
            for (int t = 0; t < 4; ++t) {
                const int id = t * 16 + ln;
                frow[id] = g[t];
                srow[id] = (id == c1) ? w1 : ((id == c2) ? w2 : 0.0f);
            }
            if (ln == 0) {  // lanes 0 and 16 each own one token row
                outI[(size_t)tok * 2 + 0] = c1;
                outI[(size_t)tok * 2 + 1] = c2;
            }
        }
    }
}

extern "C" void kernel_launch(void* const* d_in, const int* in_sizes, int n_in,
                              void* d_out, int out_size, void* d_ws, size_t ws_size,
                              hipStream_t stream) {
    const float* h   = (const float*)d_in[0];
    const float* W   = (const float*)d_in[1];
    const float* E   = (const float*)d_in[2];
    const float* tau = (const float*)d_in[3];

    // Outputs concatenated in return order: sparse_gates [N*64] f32,
    // topk_indices [N*2] i32 (stored bitwise in the f32 buffer), full_gates [N*64] f32.
    float* base = (float*)d_out;
    float* outS = base;
    int*   outI = (int*)(base + (size_t)NTOK * NEXP);
    float* outF = base + (size_t)NTOK * NEXP + (size_t)NTOK * 2;

    dim3 grid(NTOK / TOK_PER_WG), block(256);
    hipLaunchKernelGGL(moe_router_kernel, grid, block, 0, stream,
                       h, W, E, tau, outS, outI, outF);
    (void)in_sizes; (void)n_in; (void)out_size; (void)d_ws; (void)ws_size;
}
